// WaveNetModel_81990925681033
// MI455X (gfx1250) — matrix operand order repaired and checked by reading, not compiled, not run on hardware
//
#include <hip/hip_runtime.h>
#include <hip/hip_bf16.h>

// ---------------- constants (match reference) ----------------
#define LAYERS  10
#define BLOCKS  3
#define NLAY    30          // LAYERS*BLOCKS
#define DCH     128         // DC
#define RCH     128         // RC
#define SCH     256         // SC
#define NCLS    256         // CLASSES
#define LEN     8192        // L
#define ENC     16          // EN
#define BATCH   4           // N
#define SR      4096
#define TT      64          // time tile per workgroup
#define LDBR    136         // LDS row stride in halves for 128-ch tiles (256B data + 16B pad)

typedef __bf16 bf16_t;
typedef __attribute__((ext_vector_type(16))) __bf16 v16bf;
typedef __attribute__((ext_vector_type(8)))  __bf16 v8bf;
typedef __attribute__((ext_vector_type(8)))  float  v8f;
typedef unsigned int u32x4 __attribute__((ext_vector_type(4)));
typedef int          i32x4 __attribute__((ext_vector_type(4)));
typedef int          i32x8 __attribute__((ext_vector_type(8)));

#if __has_builtin(__builtin_amdgcn_tensor_load_to_lds)
#define HAVE_TDM 1
#else
#define HAVE_TDM 0
#endif

// ---------------- WMMA helpers ----------------
__device__ __forceinline__ v8f wmma_bf16(v16bf a, v16bf b, v8f c) {
  return __builtin_amdgcn_wmma_f32_16x16x32_bf16(false, a, false, b, (short)0, c, false, false);
}

__device__ __forceinline__ v16bf load_a_frag(const bf16_t* p) {
  return *(const v16bf*)p;  // 32B-aligned packed fragment
}

__device__ __forceinline__ v16bf load_b_frag(const bf16_t* p) {
  v8bf lo = *(const v8bf*)p;
  v8bf hi = *(const v8bf*)(p + 8);
  return __builtin_shufflevector(lo, hi, 0,1,2,3,4,5,6,7,8,9,10,11,12,13,14,15);
}

// ---------------- TDM: stage one [TT rows x 128ch] bf16 tile into LDS ----------------
// Tensor: 2D, data_size=2B, row = 128 elements (256B). LDS pad: +4 dwords (16B) after
// every 64 dwords (256B) -> LDS row stride 272B == LDBR halves.
#if HAVE_TDM
__device__ __forceinline__ void tdm_load_tile(void* ldsDst, const bf16_t* gsrc) {
  unsigned int lds = (unsigned int)(unsigned long long)(size_t)ldsDst; // LDS byte offset (low 32 of flat)
  unsigned long long ga = (unsigned long long)(size_t)gsrc;
  u32x4 g0;
  g0.x = 1u;                                            // count=1 (valid user descriptor)
  g0.y = lds;                                           // lds_addr
  g0.z = (unsigned int)(ga & 0xFFFFFFFFu);              // global_addr[31:0]
  g0.w = (unsigned int)((ga >> 32) & 0x01FFFFFFu)       // global_addr[56:32]
       | (2u << 30);                                    // type = 2 ("image")
  i32x8 g1;
  g1[0] = (1 << 16)      // data_size = 1 -> 2 bytes
        | (1 << 20)      // pad_enable
        | (5 << 22)      // pad_interval: 64 dwords (256B) between pads
        | (3 << 25);     // pad_amount: 4 dwords (16B)
  g1[1] = (RCH & 0xFFFF) << 16;                         // tensor_dim0[15:0] = 128
  g1[2] = ((TT & 0xFFFF) << 16);                        // tensor_dim0 hi=0 | tensor_dim1[15:0]=64
  g1[3] = (RCH << 16);                                  // tensor_dim1 hi=0 | tile_dim0=128
  g1[4] = TT;                                           // tile_dim1=64, tile_dim2=0
  g1[5] = RCH;                                          // tensor_dim0_stride = 128 elements
  g1[6] = 0;
  g1[7] = 0;
  i32x4 gz = {0, 0, 0, 0};
#if __clang_major__ >= 23
  i32x8 gz8 = {0, 0, 0, 0, 0, 0, 0, 0};
  __builtin_amdgcn_tensor_load_to_lds(g0, g1, gz, gz, gz8, 0);
#else
  __builtin_amdgcn_tensor_load_to_lds(g0, g1, gz, gz, 0);
#endif
}
#endif

// ---------------- weight packing into WMMA A-fragment order ----------------
// 16-bit A (16x32 MxK), contiguous k order used here (B loads match): lane l, half h:  m=(l&15); k=((l>=16)?16:0)+h
__global__ void pack_a_kernel(const float* __restrict__ src, bf16_t* __restrict__ dst,
                              int otiles, int ktiles,
                              long strideO, long strideK,
                              long srcLayerStride, long srcTapStride,
                              long dstLayerStride, long dstTapStride) {
  int layer = blockIdx.y;
  int tap   = blockIdx.z;
  long idx = (long)blockIdx.x * blockDim.x + threadIdx.x;
  long total = (long)otiles * ktiles * 512;
  if (idx >= total) return;
  int h    = (int)(idx & 15);
  int lane = (int)((idx >> 4) & 31);
  long frag = idx >> 9;
  int kt = (int)(frag % ktiles);
  int ot = (int)(frag / ktiles);
  int m   = ot * 16 + (lane & 15);
  int klo = (lane >= 16) ? 8 : 0;
  int k   = kt * 32 + ((lane >= 16) ? 16 : 0) + h;
  float v = src[layer * srcLayerStride + tap * srcTapStride + (long)m * strideO + (long)k * strideK];
  dst[layer * dstLayerStride + tap * dstTapStride + idx] = (bf16_t)v;
}

// ---------------- conditioning precompute ----------------
__global__ void cond_kernel(const float* __restrict__ cond_w, const float* __restrict__ en,
                            float* __restrict__ condv) {
  long idx = (long)blockIdx.x * blockDim.x + threadIdx.x;
  if (idx >= (long)NLAY * BATCH * 256 * ENC) return;
  int e = (int)(idx & 15);
  int o = (int)((idx >> 4) & 255);
  int n = (int)((idx >> 12) & 3);
  int i = (int)(idx >> 14);
  float acc = 0.f;
  const float* w = cond_w + ((long)i * (2 * DCH) + o) * NCLS;
  const float* ev = en + (long)n * NCLS * ENC + e;
  for (int c = 0; c < NCLS; ++c) acc += w[c] * ev[(long)c * ENC];
  condv[idx] = acc;
}

__global__ void endc_kernel(const float* __restrict__ end2_w, const float* __restrict__ end1_b,
                            const float* __restrict__ end2_b, const float* __restrict__ en,
                            float* __restrict__ endc) {
  long idx = (long)blockIdx.x * blockDim.x + threadIdx.x;
  if (idx >= (long)BATCH * 256 * ENC) return;
  int e = (int)(idx & 15);
  int o = (int)((idx >> 4) & 255);
  int n = (int)(idx >> 12);
  float acc = end1_b[o] + end2_b[o];
  const float* w = end2_w + (long)o * NCLS;
  const float* ev = en + (long)n * NCLS * ENC + e;
  for (int c = 0; c < NCLS; ++c) acc += w[c] * ev[(long)c * ENC];
  endc[idx] = acc;
}

// ---------------- generic 1x1-conv GEMM (start projections) ----------------
// out[n][o][t] = sum_c A[o][c]*in[n][c][t]; optional bf16 mirror [n][t][o]
template<int IC, int OT>
__global__ __launch_bounds__(256)
void gemm1x1_kernel(const float* __restrict__ in, const bf16_t* __restrict__ packA,
                    float* __restrict__ out, bf16_t* __restrict__ mir) {
  constexpr int KT  = IC / 32;
  constexpr int LDB = IC + 8;
  __shared__ bf16_t bT[TT][LDB];
  const int n  = blockIdx.y;
  const int t0 = blockIdx.x * TT;
  const int tid = threadIdx.x;
  const int t = tid & 63;
  for (int c = tid >> 6; c < IC; c += 4)
    bT[t][c] = (bf16_t)in[((long)n * IC + c) * LEN + t0 + t];
  __syncthreads();

  const int wave = tid >> 5, lane = tid & 31;
  const int col = lane & 15, hi = lane >> 4;
  for (int ot = wave; ot < OT; ot += 8) {
    #pragma unroll
    for (int ts = 0; ts < 4; ++ts) {
      v8f acc = {};
      #pragma unroll
      for (int kt = 0; kt < KT; ++kt) {
        v16bf a = load_a_frag(packA + ((long)ot * KT + kt) * 512 + lane * 16);
        v16bf b = load_b_frag(&bT[ts * 16 + col][kt * 32 + hi * 16]);
        acc = wmma_bf16(a, b, acc);
      }
      #pragma unroll
      for (int r = 0; r < 8; ++r) {
        int ch = ot * 16 + r + hi * 8;
        int tt = t0 + ts * 16 + col;
        out[((long)n * (OT * 16) + ch) * LEN + tt] = acc[r];
        if (mir) mir[((long)n * LEN + tt) * (OT * 16) + ch] = (bf16_t)acc[r];
      }
    }
  }
}

// ---------------- fused WaveNet layer ----------------
__global__ __launch_bounds__(256)
void wavenet_layer_kernel(const float* __restrict__ lin, float* __restrict__ lout,
                          const bf16_t* __restrict__ lmir, bf16_t* __restrict__ lmirOut,
                          float* __restrict__ sbuf,
                          const bf16_t* __restrict__ packDil,   // [2][16][4][512]
                          const bf16_t* __restrict__ packRes,   // [8][4][512]
                          const bf16_t* __restrict__ packSkip,  // [16][4][512]
                          const float* __restrict__ condl,      // [N][256][16]
                          int dil) {
  __shared__ bf16_t curT[TT][LDBR];
  __shared__ bf16_t delT[TT][LDBR];
  __shared__ bf16_t gT[TT][LDBR];
  __shared__ float  condS[256][ENC];

  const int n  = blockIdx.y;
  const int t0 = blockIdx.x * TT;
  const int tid = threadIdx.x;
  const int wave = tid >> 5, lane = tid & 31;

  // ---- stage cur/delayed activation tiles (bf16 mirror) into LDS ----
#if HAVE_TDM
  if (t0 >= dil) {
    if (wave == 0) {
      tdm_load_tile(&curT[0][0], lmir + ((long)n * LEN + t0) * RCH);
      tdm_load_tile(&delT[0][0], lmir + ((long)n * LEN + t0 - dil) * RCH);
      __builtin_amdgcn_s_wait_tensorcnt(0);
    }
  } else
#endif
  {
    const bf16_t* mbase = lmir + (long)n * LEN * RCH;
    int t = tid & 63;
    for (int c = tid >> 6; c < RCH; c += 4) {
      curT[t][c] = mbase[(long)(t0 + t) * RCH + c];
      int td = t0 + t - dil;
      delT[t][c] = (td >= 0) ? mbase[(long)td * RCH + c] : (bf16_t)0.f;
    }
  }
  for (int idx = tid; idx < 256 * ENC; idx += 256)
    condS[idx >> 4][idx & 15] = condl[((long)n << 12) + idx];
  __syncthreads();

  const int col = lane & 15, hi = lane >> 4;

  // ---- dilated conv + cond + gate -> gT ----
  #pragma unroll
  for (int ts = 0; ts < 4; ++ts) {
    v8f accS = {}, accT = {};
    #pragma unroll
    for (int kt = 0; kt < 4; ++kt) {
      v16bf bDel = load_b_frag(&delT[ts * 16 + col][kt * 32 + hi * 16]);
      v16bf bCur = load_b_frag(&curT[ts * 16 + col][kt * 32 + hi * 16]);
      accS = wmma_bf16(load_a_frag(packDil + ((long)((0 * 16 + wave) * 4 + kt)) * 512 + lane * 16), bDel, accS);
      accS = wmma_bf16(load_a_frag(packDil + ((long)((1 * 16 + wave) * 4 + kt)) * 512 + lane * 16), bCur, accS);
      accT = wmma_bf16(load_a_frag(packDil + ((long)((0 * 16 + wave + 8) * 4 + kt)) * 512 + lane * 16), bDel, accT);
      accT = wmma_bf16(load_a_frag(packDil + ((long)((1 * 16 + wave + 8) * 4 + kt)) * 512 + lane * 16), bCur, accT);
    }
    #pragma unroll
    for (int r = 0; r < 8; ++r) {
      int m   = r + hi * 8;
      int chS = wave * 16 + m;
      int chT = 128 + wave * 16 + m;
      float aS = accS[r] + condS[chS][col];
      float aT = accT[r] + condS[chT][col];
      float g  = (1.f / (1.f + __expf(-aS))) * tanhf(aT);
      gT[ts * 16 + col][wave * 16 + m] = (bf16_t)g;
    }
  }
  __syncthreads();

  // ---- residual: lout = lin + Wres @ g  (also refresh bf16 mirror) ----
  #pragma unroll
  for (int ts = 0; ts < 4; ++ts) {
    v8f acc = {};
    #pragma unroll
    for (int kt = 0; kt < 4; ++kt) {
      v16bf a = load_a_frag(packRes + ((long)(wave * 4 + kt)) * 512 + lane * 16);
      v16bf b = load_b_frag(&gT[ts * 16 + col][kt * 32 + hi * 16]);
      acc = wmma_bf16(a, b, acc);
    }
    #pragma unroll
    for (int r = 0; r < 8; ++r) {
      int ch = wave * 16 + r + hi * 8;
      int tt = t0 + ts * 16 + col;
      long gi = ((long)n * RCH + ch) * LEN + tt;
      float v = lin[gi] + acc[r];
      lout[gi] = v;
      lmirOut[((long)n * LEN + tt) * RCH + ch] = (bf16_t)v;
    }
  }

  // ---- skip: s += Wskip @ g ----
  #pragma unroll
  for (int oo = 0; oo < 2; ++oo) {
    int ot = wave + oo * 8;
    #pragma unroll
    for (int ts = 0; ts < 4; ++ts) {
      v8f acc = {};
      #pragma unroll
      for (int kt = 0; kt < 4; ++kt) {
        v16bf a = load_a_frag(packSkip + ((long)(ot * 4 + kt)) * 512 + lane * 16);
        v16bf b = load_b_frag(&gT[ts * 16 + col][kt * 32 + hi * 16]);
        acc = wmma_bf16(a, b, acc);
      }
      #pragma unroll
      for (int r = 0; r < 8; ++r) {
        int ch = ot * 16 + r + hi * 8;
        long gi = ((long)n * SCH + ch) * LEN + t0 + ts * 16 + col;
        sbuf[gi] += acc[r];
      }
    }
  }
}

// ---------------- end: relu(s) -> end1 GEMM + cond -> relu -> slice ----------------
__global__ __launch_bounds__(256)
void end_kernel(const float* __restrict__ sbuf, const bf16_t* __restrict__ packEnd1,
                const float* __restrict__ endc, float* __restrict__ out) {
  constexpr int LDB = SCH + 8;
  __shared__ bf16_t bT[TT][LDB];
  __shared__ float  condS[256][ENC];
  const int n  = blockIdx.y;
  const int t0 = (LEN - SR) + blockIdx.x * TT;
  const int tid = threadIdx.x;
  const int t = tid & 63;
  for (int c = tid >> 6; c < SCH; c += 4)
    bT[t][c] = (bf16_t)fmaxf(sbuf[((long)n * SCH + c) * LEN + t0 + t], 0.f);
  for (int idx = tid; idx < 256 * ENC; idx += 256)
    condS[idx >> 4][idx & 15] = endc[((long)n << 12) + idx];
  __syncthreads();

  const int wave = tid >> 5, lane = tid & 31;
  const int col = lane & 15, hi = lane >> 4;
  for (int ot = wave; ot < 16; ot += 8) {
    #pragma unroll
    for (int ts = 0; ts < 4; ++ts) {
      v8f acc = {};
      #pragma unroll
      for (int kt = 0; kt < 8; ++kt) {
        v16bf a = load_a_frag(packEnd1 + ((long)ot * 8 + kt) * 512 + lane * 16);
        v16bf b = load_b_frag(&bT[ts * 16 + col][kt * 32 + hi * 16]);
        acc = wmma_bf16(a, b, acc);
      }
      #pragma unroll
      for (int r = 0; r < 8; ++r) {
        int ch = ot * 16 + r + hi * 8;
        float v = fmaxf(acc[r] + condS[ch][col], 0.f);
        out[((long)n * NCLS + ch) * SR + (t0 - (LEN - SR)) + ts * 16 + col] = v;
      }
    }
  }
}

// ---------------- host launcher ----------------
extern "C" void kernel_launch(void* const* d_in, const int* in_sizes, int n_in,
                              void* d_out, int out_size, void* d_ws, size_t ws_size,
                              hipStream_t stream) {
  const float* x        = (const float*)d_in[0];
  const float* en       = (const float*)d_in[1];
  const float* start1_w = (const float*)d_in[2];
  const float* start2_w = (const float*)d_in[3];
  const float* cond_w   = (const float*)d_in[4];
  const float* dil_w    = (const float*)d_in[5];
  const float* res_w    = (const float*)d_in[6];
  const float* skip_w   = (const float*)d_in[7];
  const float* end1_w   = (const float*)d_in[8];
  const float* end1_b   = (const float*)d_in[9];
  const float* end2_w   = (const float*)d_in[10];
  const float* end2_b   = (const float*)d_in[11];
  float* out = (float*)d_out;
  (void)in_sizes; (void)n_in; (void)out_size; (void)ws_size;

  char* ws = (char*)d_ws;
  size_t off = 0;
  auto take = [&](size_t bytes) -> char* {
    char* p = ws + off;
    off += (bytes + 255) & ~(size_t)255;
    return p;
  };
  bf16_t* packDil  = (bf16_t*)take((size_t)NLAY * 2 * 16 * 4 * 512 * 2);
  bf16_t* packRes  = (bf16_t*)take((size_t)NLAY * 8  * 4 * 512 * 2);
  bf16_t* packSkip = (bf16_t*)take((size_t)NLAY * 16 * 4 * 512 * 2);
  bf16_t* packS1   = (bf16_t*)take((size_t)8  * 8 * 512 * 2);
  bf16_t* packS2   = (bf16_t*)take((size_t)16 * 4 * 512 * 2);
  bf16_t* packE1   = (bf16_t*)take((size_t)16 * 8 * 512 * 2);
  float*  condv    = (float*)take((size_t)NLAY * BATCH * 256 * ENC * 4);
  float*  endc     = (float*)take((size_t)BATCH * 256 * ENC * 4);
  float*  lA       = (float*)take((size_t)BATCH * RCH * LEN * 4);
  float*  lB       = (float*)take((size_t)BATCH * RCH * LEN * 4);
  bf16_t* lmA      = (bf16_t*)take((size_t)BATCH * LEN * RCH * 2);  // bf16 mirror [n][t][ch]
  bf16_t* lmB      = (bf16_t*)take((size_t)BATCH * LEN * RCH * 2);
  float*  sbuf     = (float*)take((size_t)BATCH * SCH * LEN * 4);

  // --- pack weights into WMMA fragment order ---
  pack_a_kernel<<<dim3(128, NLAY, 2), 256, 0, stream>>>(
      dil_w, packDil, 16, 4, 256, 2,
      (long)256 * 128 * 2, 1, (long)2 * 16 * 4 * 512, (long)16 * 4 * 512);
  pack_a_kernel<<<dim3(64, NLAY, 1), 256, 0, stream>>>(
      res_w, packRes, 8, 4, 128, 1, (long)128 * 128, 0, (long)8 * 4 * 512, 0);
  pack_a_kernel<<<dim3(128, NLAY, 1), 256, 0, stream>>>(
      skip_w, packSkip, 16, 4, 128, 1, (long)256 * 128, 0, (long)16 * 4 * 512, 0);
  pack_a_kernel<<<dim3(128, 1, 1), 256, 0, stream>>>(
      start1_w, packS1, 8, 8, 256, 1, 0, 0, 0, 0);
  pack_a_kernel<<<dim3(128, 1, 1), 256, 0, stream>>>(
      start2_w, packS2, 16, 4, 128, 1, 0, 0, 0, 0);
  pack_a_kernel<<<dim3(256, 1, 1), 256, 0, stream>>>(
      end1_w, packE1, 16, 8, 256, 1, 0, 0, 0, 0);

  // --- conditioning precompute ---
  cond_kernel<<<dim3((NLAY * BATCH * 256 * ENC + 255) / 256), 256, 0, stream>>>(cond_w, en, condv);
  endc_kernel<<<dim3((BATCH * 256 * ENC + 255) / 256), 256, 0, stream>>>(end2_w, end1_b, end2_b, en, endc);

  // --- start projections (start1 also emits the bf16 mirror) ---
  gemm1x1_kernel<NCLS, RCH / 16><<<dim3(LEN / TT, BATCH), 256, 0, stream>>>(x, packS1, lA, lmA);
  gemm1x1_kernel<RCH, SCH / 16><<<dim3(LEN / TT, BATCH), 256, 0, stream>>>(lA, packS2, sbuf, (bf16_t*)nullptr);

  // --- fused layers (double-buffered l + bf16 mirror) ---
  float* cur = lA;   float* nxt = lB;
  bf16_t* mcur = lmA; bf16_t* mnxt = lmB;
  for (int i = 0; i < NLAY; ++i) {
    int dil = 1 << (i % LAYERS);
    wavenet_layer_kernel<<<dim3(LEN / TT, BATCH), 256, 0, stream>>>(
        cur, nxt, mcur, mnxt, sbuf,
        packDil + (size_t)i * 2 * 16 * 4 * 512,
        packRes + (size_t)i * 8 * 4 * 512,
        packSkip + (size_t)i * 16 * 4 * 512,
        condv + (size_t)i * BATCH * 256 * ENC,
        dil);
    float* tf = cur; cur = nxt; nxt = tf;
    bf16_t* tm = mcur; mcur = mnxt; mnxt = tm;
  }

  // --- end projection + slice ---
  end_kernel<<<dim3(SR / TT, BATCH), 256, 0, stream>>>(sbuf, packE1, endc, out);
}
